// MambaFusion_2010044695186
// MI455X (gfx1250) — matrix operand order repaired and checked by reading, not compiled, not run on hardware
//
#include <hip/hip_runtime.h>

typedef __attribute__((ext_vector_type(16))) __bf16        v16bf;
typedef __attribute__((ext_vector_type(8)))  float         v8f;
typedef __attribute__((ext_vector_type(4)))  unsigned int  v4u;
typedef unsigned short ushort_t;

#define DIMC     512
#define D_STATE  16
#define D_INNER  1024
#define DT_RANK  32
#define NLAYERS  3
#define PREFIX   64
#define BATCH    8
#define SEQ_N    1024
#define S0LEN    1088
#define L_TOT    3072            /* NLAYERS * SEQ_N        */
#define TOKENS   24576           /* BATCH * L_TOT          */
#define NCHUNK   24              /* scan chunks            */
#define LCHUNK   128             /* steps per chunk        */

__device__ __forceinline__ ushort_t f32_to_bf16(float f){
  unsigned int u = __float_as_uint(f);
  u += 0x7FFFu + ((u >> 16) & 1u);        // round-to-nearest-even
  return (ushort_t)(u >> 16);
}
__device__ __forceinline__ float bf16_to_f32(ushort_t h){
  return __uint_as_float(((unsigned int)h) << 16);
}

__device__ __forceinline__ float block_sum_256(float v, float* red){
  int tid = threadIdx.x;
  red[tid] = v; __syncthreads();
  #pragma unroll
  for (int s = 128; s > 0; s >>= 1){
    if (tid < s) red[tid] += red[tid + s];
    __syncthreads();
  }
  float r = red[0]; __syncthreads();
  return r;
}

/* ------------------------- f32 -> bf16 convert ------------------------- */
__global__ void cvt_bf16_kernel(const float* __restrict__ in, ushort_t* __restrict__ out, int n){
  int i = blockIdx.x * 256 + threadIdx.x;
  if (i < n) out[i] = f32_to_bf16(in[i]);
}

/* --------------------- LN1: gather + layernorm -> bf16 ------------------ */
__global__ __launch_bounds__(256) void ln1_kernel(const float* __restrict__ feats,
                                                  const float* __restrict__ g,
                                                  const float* __restrict__ b,
                                                  ushort_t* __restrict__ xln){
  int t  = blockIdx.x;                 // token 0..TOKENS-1
  int bb = t / L_TOT, r = t % L_TOT, l = r >> 10, n = r & 1023;
  const float* row = feats + (((size_t)l * BATCH + bb) * S0LEN + PREFIX + n) * DIMC;
  __shared__ float red[256];
  int i0 = threadIdx.x, i1 = threadIdx.x + 256;
  float v0 = row[i0], v1 = row[i1];
  float mu  = block_sum_256(v0 + v1, red) * (1.0f / DIMC);
  float d0 = v0 - mu, d1 = v1 - mu;
  float var = block_sum_256(d0 * d0 + d1 * d1, red) * (1.0f / DIMC);
  float inv = rsqrtf(var + 1e-5f);
  ushort_t* o = xln + (size_t)t * DIMC;
  o[i0] = f32_to_bf16(d0 * inv * g[i0] + b[i0]);
  o[i1] = f32_to_bf16(d1 * inv * g[i1] + b[i1]);
}

/* ----------------------------- WMMA GEMM -------------------------------- */
/* C[M,N] = A[M,K] @ W[N,K]^T ; A,W bf16 row-major ; f32 accumulate.
   Per wave: one 32x32 tile (2x2 WMMA accumulators). Modes select epilogue. */
union FragU { v16bf v; v4u q[2]; };

__device__ __forceinline__ void epi_store(int mode, int m, int n, int N, float val,
                                          float* __restrict__ o0, ushort_t* __restrict__ o1,
                                          const float* __restrict__ bias){
  if (mode == 0) {                                   // plain f32
    o0[(size_t)m * N + n] = val;
  } else if (mode == 1) {                            // xz split: xi f32 / silu(z) bf16
    if (n < D_INNER) o0[(size_t)m * D_INNER + n] = val;
    else {
      float s = val / (1.0f + __expf(-val));
      o1[(size_t)m * D_INNER + (n - D_INNER)] = f32_to_bf16(s);
    }
  } else if (mode == 2) {                            // dbc f32 + dt_raw bf16 (n<32)
    o0[(size_t)m * 64 + n] = val;
    if (n < DT_RANK) o1[(size_t)m * DT_RANK + n] = f32_to_bf16(val);
  } else {                                           // dt = softplus(val + bias[n])
    float x  = val + bias[n];
    float sp = (x > 20.0f) ? x : log1pf(__expf(x));
    o0[(size_t)m * D_INNER + n] = sp;
  }
}

__global__ __launch_bounds__(256) void gemm_bf16_kernel(
    const ushort_t* __restrict__ A, const ushort_t* __restrict__ W,
    int M, int N, int K, int mode,
    float* __restrict__ o0, ushort_t* __restrict__ o1,
    const float* __restrict__ bias)
{
  int tilesN = N >> 5;
  int wave = (int)blockIdx.x * 8 + ((int)threadIdx.x >> 5);
  int lane = (int)threadIdx.x & 31;
  if (wave >= (M >> 5) * tilesN) return;             // wave-uniform guard
  int tm = wave / tilesN, tn = wave % tilesN;
  int m0 = tm << 5, n0 = tn << 5;

  int rA0 = m0 + (lane & 15), rA1 = rA0 + 16;
  int rB0 = n0 + (lane & 15), rB1 = rB0 + 16;
  int aOff = (lane >> 4) << 3;                       // A: 0 / 8 element chunk base
  int bOff = (lane >> 4) << 3;                       // B: 0 / 8 element chunk base

  v8f acc00 = {}, acc01 = {}, acc10 = {}, acc11 = {};
  for (int k = 0; k < K; k += 32) {
    FragU a0, a1, b0, b1;
    const ushort_t* pa0 = A + (size_t)rA0 * K + k + aOff;
    const ushort_t* pa1 = A + (size_t)rA1 * K + k + aOff;
    a0.q[0] = *(const v4u*)(pa0);  a0.q[1] = *(const v4u*)(pa0 + 16);
    a1.q[0] = *(const v4u*)(pa1);  a1.q[1] = *(const v4u*)(pa1 + 16);
    const ushort_t* pb0 = W + (size_t)rB0 * K + k + bOff;
    const ushort_t* pb1 = W + (size_t)rB1 * K + k + bOff;
    b0.q[0] = *(const v4u*)(pb0);  b0.q[1] = *(const v4u*)(pb0 + 16);
    b1.q[0] = *(const v4u*)(pb1);  b1.q[1] = *(const v4u*)(pb1 + 16);
    acc00 = __builtin_amdgcn_wmma_f32_16x16x32_bf16(false, a0.v, false, b0.v, (short)0, acc00, false, false);
    acc01 = __builtin_amdgcn_wmma_f32_16x16x32_bf16(false, a0.v, false, b1.v, (short)0, acc01, false, false);
    acc10 = __builtin_amdgcn_wmma_f32_16x16x32_bf16(false, a1.v, false, b0.v, (short)0, acc10, false, false);
    acc11 = __builtin_amdgcn_wmma_f32_16x16x32_bf16(false, a1.v, false, b1.v, (short)0, acc11, false, false);
  }
  int lm = (lane >> 4) << 3;
  int nc = lane & 15;
  #pragma unroll
  for (int v = 0; v < 8; v++){
    epi_store(mode, m0 + lm + v,      n0 + nc,      N, acc00[v], o0, o1, bias);
    epi_store(mode, m0 + lm + v,      n0 + 16 + nc, N, acc01[v], o0, o1, bias);
    epi_store(mode, m0 + 16 + lm + v, n0 + nc,      N, acc10[v], o0, o1, bias);
    epi_store(mode, m0 + 16 + lm + v, n0 + 16 + nc, N, acc11[v], o0, o1, bias);
  }
}

/* ------------------- causal depthwise conv (k=4) + silu ------------------ */
__global__ void conv_kernel(const float* __restrict__ xi, const float* __restrict__ cw,
                            const float* __restrict__ cb, float* __restrict__ xc,
                            ushort_t* __restrict__ xcb){
  size_t idx = (size_t)blockIdx.x * 256 + threadIdx.x;   // token*1024 + d
  int d   = (int)(idx & 1023);
  int tok = (int)(idx >> 10);
  int bb = tok / L_TOT, t = tok % L_TOT;
  float s = cb[d];
  #pragma unroll
  for (int k = 0; k < 4; k++){
    int ts = t + k - 3;
    if (ts >= 0) s += xi[((size_t)(bb * L_TOT + ts)) * D_INNER + d] * cw[d * 4 + k];
  }
  float v = s / (1.0f + __expf(-s));                     // silu
  xc[idx]  = v;
  xcb[idx] = f32_to_bf16(v);
}

/* --------------- chunked selective scan: pass 1 (local) ------------------ */
/* block -> (b, chunk, dblock); computes chunk-local h_end (from h=0) and the
   per-state decay product P = prod_t exp(dt_t * a_s) over the chunk.        */
__global__ __launch_bounds__(256) void scan_local_kernel(
    const float* __restrict__ dbc, const float* __restrict__ dtp,
    const float* __restrict__ xcp, const float* __restrict__ A_log,
    float* __restrict__ hfin, float* __restrict__ pprod)
{
  int bc = blockIdx.x >> 2;                 // b*NCHUNK + c
  int bb = bc / NCHUNK, c = bc % NCHUNK;
  int d  = ((blockIdx.x & 3) << 8) + threadIdx.x;
  int t0 = c * LCHUNK;

  __shared__ float bt[LCHUNK][D_STATE];     // B_t tile: 8 KB
  for (int i = threadIdx.x; i < LCHUNK * D_STATE; i += 256){
    int tt = i >> 4, s = i & 15;
    bt[tt][s] = dbc[((size_t)(bb * L_TOT + t0 + tt)) * 64 + DT_RANK + s];
  }
  __syncthreads();

  float a[D_STATE], h[D_STATE], P[D_STATE];
  #pragma unroll
  for (int s = 0; s < D_STATE; s++){
    a[s] = -__expf(A_log[d * D_STATE + s]); h[s] = 0.0f; P[s] = 1.0f;
  }
  for (int tt = 0; tt < LCHUNK; tt++){
    size_t tok = (size_t)bb * L_TOT + t0 + tt;
    float dt  = dtp[tok * D_INNER + d];
    float x   = xcp[tok * D_INNER + d];
    float dtx = dt * x;
    #pragma unroll
    for (int s = 0; s < D_STATE; s++){
      float e = __expf(dt * a[s]);
      P[s] *= e;
      h[s] = e * h[s] + dtx * bt[tt][s];
    }
  }
  size_t base = (((size_t)bc) * D_INNER + d) * D_STATE;
  #pragma unroll
  for (int s = 0; s < D_STATE; s++){ hfin[base + s] = h[s]; pprod[base + s] = P[s]; }
}

/* ------------- chunked scan: pass 2 (prefix across chunks) --------------- */
/* h_start[c] = h ; h = P[c]*h + h_end[c]  (exact by linearity)              */
__global__ __launch_bounds__(256) void scan_prefix_kernel(
    const float* __restrict__ hfin, const float* __restrict__ pprod,
    float* __restrict__ hstart)
{
  int bb = blockIdx.x >> 2;
  int d  = ((blockIdx.x & 3) << 8) + threadIdx.x;
  float h[D_STATE];
  #pragma unroll
  for (int s = 0; s < D_STATE; s++) h[s] = 0.0f;
  for (int c = 0; c < NCHUNK; c++){
    size_t base = (((size_t)(bb * NCHUNK + c)) * D_INNER + d) * D_STATE;
    #pragma unroll
    for (int s = 0; s < D_STATE; s++){
      hstart[base + s] = h[s];
      h[s] = pprod[base + s] * h[s] + hfin[base + s];
    }
  }
}

/* ------- chunked scan: pass 3 (re-scan with true h_start, emit y) -------- */
__global__ __launch_bounds__(256) void scan_emit_kernel(
    const float* __restrict__ dbc, const float* __restrict__ dtp,
    const float* __restrict__ xcp, const ushort_t* __restrict__ zsil,
    const float* __restrict__ A_log, const float* __restrict__ Dp,
    const float* __restrict__ hstart, ushort_t* __restrict__ ycomb)
{
  int bc = blockIdx.x >> 2;
  int bb = bc / NCHUNK, c = bc % NCHUNK;
  int d  = ((blockIdx.x & 3) << 8) + threadIdx.x;
  int t0 = c * LCHUNK;

  __shared__ float bct[LCHUNK][2 * D_STATE];   // B_t | C_t tile: 16 KB
  for (int i = threadIdx.x; i < LCHUNK * 32; i += 256){
    int tt = i >> 5, j = i & 31;
    bct[tt][j] = dbc[((size_t)(bb * L_TOT + t0 + tt)) * 64 + DT_RANK + j];
  }
  __syncthreads();

  float a[D_STATE], h[D_STATE];
  size_t base = (((size_t)bc) * D_INNER + d) * D_STATE;
  #pragma unroll
  for (int s = 0; s < D_STATE; s++){
    a[s] = -__expf(A_log[d * D_STATE + s]);
    h[s] = hstart[base + s];
  }
  float dpd = Dp[d];
  for (int tt = 0; tt < LCHUNK; tt++){
    size_t tok = (size_t)bb * L_TOT + t0 + tt;
    float dt  = dtp[tok * D_INNER + d];
    float x   = xcp[tok * D_INNER + d];
    float dtx = dt * x;
    float acc = 0.0f;
    #pragma unroll
    for (int s = 0; s < D_STATE; s++){
      float e = __expf(dt * a[s]);
      h[s] = e * h[s] + dtx * bct[tt][s];
      acc += h[s] * bct[tt][D_STATE + s];
    }
    float z = bf16_to_f32(zsil[tok * D_INNER + d]);
    ycomb[tok * D_INNER + d] = f32_to_bf16((acc + x * dpd) * z);
  }
}

/* ----------------- norm2: residual add + layernorm -> f32 ---------------- */
__global__ __launch_bounds__(256) void norm2_kernel(const float* __restrict__ mo,
                                                    const float* __restrict__ feats,
                                                    const float* __restrict__ g,
                                                    const float* __restrict__ b,
                                                    float* __restrict__ x2){
  int t  = blockIdx.x;
  int bb = t / L_TOT, r = t % L_TOT, l = r >> 10, n = r & 1023;
  const float* res = feats + (((size_t)l * BATCH + bb) * S0LEN + PREFIX + n) * DIMC;
  const float* mr  = mo + (size_t)t * DIMC;
  __shared__ float red[256];
  int i0 = threadIdx.x, i1 = threadIdx.x + 256;
  float v0 = mr[i0] + res[i0], v1 = mr[i1] + res[i1];
  float mu  = block_sum_256(v0 + v1, red) * (1.0f / DIMC);
  float d0 = v0 - mu, d1 = v1 - mu;
  float var = block_sum_256(d0 * d0 + d1 * d1, red) * (1.0f / DIMC);
  float inv = rsqrtf(var + 1e-5f);
  float* o = x2 + (size_t)t * DIMC;
  o[i0] = d0 * inv * g[i0] + b[i0];
  o[i1] = d1 * inv * g[i1] + b[i1];
}

/* -------------------- layer-sum + final layernorm ------------------------ */
__global__ __launch_bounds__(256) void finalln_kernel(const float* __restrict__ x2,
                                                      const float* __restrict__ g,
                                                      const float* __restrict__ b,
                                                      float* __restrict__ out){
  int blk = blockIdx.x;                                  // 0..8191 : (b, n)
  int bb = blk >> 10, n = blk & 1023;
  __shared__ float red[256];
  int i0 = threadIdx.x, i1 = threadIdx.x + 256;
  float v0 = 0.0f, v1 = 0.0f;
  #pragma unroll
  for (int l = 0; l < NLAYERS; l++){
    const float* r = x2 + ((size_t)bb * L_TOT + l * SEQ_N + n) * DIMC;
    v0 += r[i0]; v1 += r[i1];
  }
  float mu  = block_sum_256(v0 + v1, red) * (1.0f / DIMC);
  float d0 = v0 - mu, d1 = v1 - mu;
  float var = block_sum_256(d0 * d0 + d1 * d1, red) * (1.0f / DIMC);
  float inv = rsqrtf(var + 1e-5f);
  float* o = out + ((size_t)bb * SEQ_N + n) * DIMC;
  o[i0] = d0 * inv * g[i0] + b[i0];
  o[i1] = d1 * inv * g[i1] + b[i1];
}

/* ------------------------------- launch ---------------------------------- */
extern "C" void kernel_launch(void* const* d_in, const int* in_sizes, int n_in,
                              void* d_out, int out_size, void* d_ws, size_t ws_size,
                              hipStream_t stream)
{
  (void)in_sizes; (void)n_in; (void)out_size; (void)ws_size;
  const float* feats  = (const float*)d_in[0];
  const float* n1g    = (const float*)d_in[1];
  const float* n1b    = (const float*)d_in[2];
  const float* n2g    = (const float*)d_in[3];
  const float* n2b    = (const float*)d_in[4];
  const float* W_in   = (const float*)d_in[5];
  const float* conv_w = (const float*)d_in[6];
  const float* conv_b = (const float*)d_in[7];
  const float* W_x    = (const float*)d_in[8];
  const float* W_dt   = (const float*)d_in[9];
  const float* b_dt   = (const float*)d_in[10];
  const float* A_log  = (const float*)d_in[11];
  const float* Dp     = (const float*)d_in[12];
  const float* W_out  = (const float*)d_in[13];
  const float* nfg    = (const float*)d_in[14];
  const float* nfb    = (const float*)d_in[15];

  char* ws = (char*)d_ws;
  size_t off = 0;
  auto carve = [&](size_t bytes)->char*{
    char* p = ws + off; off += (bytes + 255) & ~(size_t)255; return p;
  };
  ushort_t* wInB  = (ushort_t*)carve((size_t)2048 * 512  * 2);
  ushort_t* wXB   = (ushort_t*)carve((size_t)64   * 1024 * 2);
  ushort_t* wDtB  = (ushort_t*)carve((size_t)1024 * 32   * 2);
  ushort_t* wOutB = (ushort_t*)carve((size_t)512  * 1024 * 2);
  ushort_t* xln   = (ushort_t*)carve((size_t)TOKENS * DIMC    * 2);
  float*    xi    = (float*)   carve((size_t)TOKENS * D_INNER * 4);
  ushort_t* zsil  = (ushort_t*)carve((size_t)TOKENS * D_INNER * 2);
  float*    xc    = (float*)   carve((size_t)TOKENS * D_INNER * 4);
  ushort_t* xcb   = (ushort_t*)carve((size_t)TOKENS * D_INNER * 2);
  float*    dbc   = (float*)   carve((size_t)TOKENS * 64      * 4);
  ushort_t* dtr   = (ushort_t*)carve((size_t)TOKENS * DT_RANK * 2);
  float*    dtf   = (float*)   carve((size_t)TOKENS * D_INNER * 4);
  ushort_t* ycomb = (ushort_t*)carve((size_t)TOKENS * D_INNER * 2);
  float*    mo    = (float*)   carve((size_t)TOKENS * DIMC    * 4);
  float*    x2    = (float*)   carve((size_t)TOKENS * DIMC    * 4);
  size_t    hsz   = (size_t)BATCH * NCHUNK * D_INNER * D_STATE * 4;
  float*    hfin  = (float*)   carve(hsz);
  float*    pprod = (float*)   carve(hsz);
  float*    hst   = (float*)   carve(hsz);

  // weight converts
  cvt_bf16_kernel<<<(2048 * 512 + 255) / 256, 256, 0, stream>>>(W_in,  wInB,  2048 * 512);
  cvt_bf16_kernel<<<(64 * 1024 + 255) / 256, 256, 0, stream>>>(W_x,   wXB,   64 * 1024);
  cvt_bf16_kernel<<<(1024 * 32 + 255) / 256, 256, 0, stream>>>(W_dt,  wDtB,  1024 * 32);
  cvt_bf16_kernel<<<(512 * 1024 + 255) / 256, 256, 0, stream>>>(W_out, wOutB, 512 * 1024);

  // 1) gather + LN1 -> bf16
  ln1_kernel<<<TOKENS, 256, 0, stream>>>(feats, n1g, n1b, xln);
  // 2) xz = xln @ W_in^T ; split xi (f32) / silu(z) (bf16)
  gemm_bf16_kernel<<<(TOKENS / 32) * (2048 / 32) / 8, 256, 0, stream>>>(
      xln, wInB, TOKENS, 2048, 512, 1, xi, zsil, nullptr);
  // 3) causal depthwise conv + silu -> xc (f32 + bf16)
  conv_kernel<<<(TOKENS * D_INNER) / 256, 256, 0, stream>>>(xi, conv_w, conv_b, xc, xcb);
  // 4) dbc = xc @ W_x^T (f32) ; dt_raw -> bf16
  gemm_bf16_kernel<<<(TOKENS / 32) * (64 / 32) / 8, 256, 0, stream>>>(
      xcb, wXB, TOKENS, 64, 1024, 2, dbc, dtr, nullptr);
  // 5) dt = softplus(dt_raw @ W_dt^T + b_dt)
  gemm_bf16_kernel<<<(TOKENS / 32) * (1024 / 32) / 8, 256, 0, stream>>>(
      dtr, wDtB, TOKENS, 1024, 32, 3, dtf, nullptr, b_dt);
  // 6) chunked selective scan (3 passes), fused (y + xc*Dp)*silu(z) -> bf16
  scan_local_kernel<<<BATCH * NCHUNK * 4, 256, 0, stream>>>(dbc, dtf, xc, A_log, hfin, pprod);
  scan_prefix_kernel<<<BATCH * 4, 256, 0, stream>>>(hfin, pprod, hst);
  scan_emit_kernel<<<BATCH * NCHUNK * 4, 256, 0, stream>>>(dbc, dtf, xc, zsil, A_log, Dp, hst, ycomb);
  // 7) m = ycomb @ W_out^T
  gemm_bf16_kernel<<<(TOKENS / 32) * (512 / 32) / 8, 256, 0, stream>>>(
      ycomb, wOutB, TOKENS, 512, 1024, 0, mo, nullptr, nullptr);
  // 8) x2 = LN2(m + residual)
  norm2_kernel<<<TOKENS, 256, 0, stream>>>(mo, feats, n2g, n2b, x2);
  // 9) out = LNf(sum over layers)
  finalln_kernel<<<BATCH * SEQ_N, 256, 0, stream>>>(x2, nfg, nfb, (float*)d_out);
}
